// matchLSTM_7344394076365
// MI455X (gfx1250) — matrix operand order repaired and checked by reading, not compiled, not run on hardware
//
#include <hip/hip_runtime.h>

// ---------------------------------------------------------------------------
// matchLSTM for MI455X (gfx1250): bf16 WMMA GEMMs with fused LSTM epilogues.
// ---------------------------------------------------------------------------

typedef __bf16 bf16_t;
typedef __attribute__((ext_vector_type(16))) __bf16 v16bf;
typedef __attribute__((ext_vector_type(8)))  __bf16 v8bf;
typedef __attribute__((ext_vector_type(8)))  float   v8f;

#define DEV static __device__ __forceinline__

#define BB 128   // batch
#define TP 128   // premise length
#define THL 64   // hypothesis length
#define EE 300   // embed dim
#define EP 320   // embed dim padded to mult of 32
#define HH 512   // hidden
#define G4 2048  // 4*H

// ---------------- scalar helpers ----------------
DEV float bf2f(bf16_t x) {
  unsigned short u = __builtin_bit_cast(unsigned short, x);
  unsigned int v = ((unsigned int)u) << 16;
  return __builtin_bit_cast(float, v);
}
DEV bf16_t f2bf(float f) {
  unsigned int u = __builtin_bit_cast(unsigned int, f);
  unsigned int r = (u + 0x7FFFu + ((u >> 16) & 1u)) >> 16;
  return __builtin_bit_cast(bf16_t, (unsigned short)r);
}
DEV float fast_sig(float x) { return 1.0f / (1.0f + __expf(-x)); }
DEV float fast_tanh(float x) {
  x = fminf(fmaxf(x, -15.0f), 15.0f);
  float e = __expf(2.0f * x);
  return (e - 1.0f) / (e + 1.0f);
}

// ---------------- WMMA fragment loaders ----------------
// A (16x32 MxK, bf16): lane<16 holds row m0+lane, K chunks {k0..k0+7, k0+16..k0+23}
//                      lane>=16 holds row m0+lane-16, chunks {k0+8.., k0+24..}
DEV v16bf load_frag_a(const bf16_t* rowp, int k0, int lane) {
  const int off = (lane & 16) ? 8 : 0;
  v8bf lo = *(const v8bf*)(rowp + k0 + off);
  v8bf hi = *(const v8bf*)(rowp + k0 + 16 + off);
  v16bf r;
#pragma unroll
  for (int i = 0; i < 8; ++i) { r[i] = lo[i]; r[i + 8] = hi[i]; }
  return r;
}
// B (32x16 KxN, bf16) with weights stored [n][k] row-major (k contiguous):
// lane holds column n0+lane%16, K chunks {k0..k0+7, k0+16..k0+23} (lanes<16)
// or {k0+8..k0+15, k0+24..k0+31}.
DEV v16bf load_frag_b(const bf16_t* rowp, int k0, int lane) {
  const int off = (lane & 16) ? 8 : 0;
  v8bf lo = *(const v8bf*)(rowp + k0 + off);
  v8bf hi = *(const v8bf*)(rowp + k0 + 16 + off);
  v16bf r;
#pragma unroll
  for (int i = 0; i < 8; ++i) { r[i] = lo[i]; r[i + 8] = hi[i]; }
  return r;
}
DEV v8f wmma_bf16(v16bf a, v16bf b, v8f c) {
  return __builtin_amdgcn_wmma_f32_16x16x32_bf16(false, a, false, b, (short)0, c,
                                                 false, false);
}

// Accumulate one A-source against the 4 gate weight rows.
template <int KLEN>
DEV void mm4(const bf16_t* ap, const bf16_t* const* wp, int kc, int lane, v8f* acc) {
  for (int k0 = 0; k0 < KLEN; k0 += 32) {
    const v16bf af = load_frag_a(ap, k0, lane);
#pragma unroll
    for (int g = 0; g < 4; ++g) {
      const v16bf bfr = load_frag_b(wp[g], kc + k0, lane);
      acc[g] = wmma_bf16(af, bfr, acc[g]);
    }
  }
}

// ---------------------------------------------------------------------------
// Fused LSTM/LSTMCell step:  gates = [A0|A1|A2] @ Wcat^T + bias, then the
// pointwise LSTM update.  Each wave owns one 16x16 (batch x hidden) tile and
// computes all four gate quadrants so the epilogue needs no cross-wave data.
// grid = 32 blocks x 256 threads = 256 waves = 8 Mtiles x 32 Ntiles.
// ---------------------------------------------------------------------------
template <int K0, int K1, int K2>
__global__ __launch_bounds__(256) void rnn_step_kernel(
    const bf16_t* __restrict__ a0, long s0,
    const bf16_t* __restrict__ a1, long s1,
    const bf16_t* __restrict__ a2, long s2,
    const bf16_t* __restrict__ W,     // [2048][K0+K1+K2] bf16, n-major
    const float* __restrict__ bias,   // [2048]  (bih+bhh)
    float* __restrict__ cst,          // [128][512] cell state (fp32)
    const bf16_t* __restrict__ hprev, // [128][512] previous h (bf16)
    bf16_t* __restrict__ hout,        // [128][512] new h (bf16)
    bf16_t* __restrict__ seq, long seq_s,  // optional per-t output (masked->0)
    const int* __restrict__ lens, int t) {
  constexpr int KT = K0 + K1 + K2;
  const int lane = threadIdx.x & 31;
  const int wave = threadIdx.x >> 5;
  const int tile = blockIdx.x * 8 + wave;
  const int m0 = (tile & 7) * 16;
  const int n0 = (tile >> 3) * 16;
  const int l15 = lane & 15;
  const int arow = m0 + l15;
  const int ncol = n0 + l15;

  const bf16_t* wp[4];
#pragma unroll
  for (int g = 0; g < 4; ++g) wp[g] = W + (long)(g * HH + ncol) * KT;

  v8f acc[4];
#pragma unroll
  for (int g = 0; g < 4; ++g)
#pragma unroll
    for (int r = 0; r < 8; ++r) acc[g][r] = 0.0f;

  mm4<K0>(a0 + (long)arow * s0, wp, 0, lane, acc);
  if constexpr (K1 > 0) mm4<K1>(a1 + (long)arow * s1, wp, K0, lane, acc);
  if constexpr (K2 > 0) mm4<K2>(a2 + (long)arow * s2, wp, K0 + K1, lane, acc);

  const float bi = bias[0 * HH + ncol];
  const float bf_ = bias[1 * HH + ncol];
  const float bg = bias[2 * HH + ncol];
  const float bo = bias[3 * HH + ncol];

#pragma unroll
  for (int r = 0; r < 8; ++r) {
    const int row = m0 + r + ((lane & 16) ? 8 : 0);
    const long idx = (long)row * HH + ncol;
    const float gi = acc[0][r] + bi;
    const float gf = acc[1][r] + bf_;
    const float gg = acc[2][r] + bg;
    const float go = acc[3][r] + bo;
    const float cold = cst[idx];
    const float cn = fast_sig(gf) * cold + fast_sig(gi) * fast_tanh(gg);
    const float hn = fast_sig(go) * fast_tanh(cn);
    const bool mk = t < lens[row];
    cst[idx] = mk ? cn : cold;
    hout[idx] = mk ? f2bf(hn) : hprev[idx];
    if (seq) seq[(long)row * seq_s + ncol] = f2bf(mk ? hn : 0.0f);
  }
}

// ---------------------------------------------------------------------------
// Generic bf16 GEMM:  out[m][n] = sum_k A[m][k] * W[n][k]   (up to 2 A sources)
// Used for Ws_pre = h_s @ W_s^T  (M=16384) and q = [htk|h_m] @ [W_t|W_m]^T.
// ---------------------------------------------------------------------------
template <int K0, int K1, bool OBF>
__global__ __launch_bounds__(256) void gemm_kernel(
    const bf16_t* __restrict__ a0, long s0,
    const bf16_t* __restrict__ a1, long s1,
    const bf16_t* __restrict__ W,  // [N][K0+K1]
    void* __restrict__ outp, int M, int N) {
  constexpr int KT = K0 + K1;
  const int lane = threadIdx.x & 31;
  const int wave = threadIdx.x >> 5;
  const int tile = blockIdx.x * 8 + wave;
  const int mtiles = M >> 4;
  const int mt = tile % mtiles;
  const int nt = tile / mtiles;
  const int m0 = mt * 16, n0 = nt * 16;
  const int l15 = lane & 15;
  const int arow = m0 + l15, ncol = n0 + l15;
  const bf16_t* wr = W + (long)ncol * KT;

  v8f acc;
#pragma unroll
  for (int r = 0; r < 8; ++r) acc[r] = 0.0f;

  const bf16_t* p0 = a0 + (long)arow * s0;
  for (int k0 = 0; k0 < K0; k0 += 32)
    acc = wmma_bf16(load_frag_a(p0, k0, lane), load_frag_b(wr, k0, lane), acc);
  if constexpr (K1 > 0) {
    const bf16_t* p1 = a1 + (long)arow * s1;
    for (int k0 = 0; k0 < K1; k0 += 32)
      acc = wmma_bf16(load_frag_a(p1, k0, lane), load_frag_b(wr, K0 + k0, lane), acc);
  }
#pragma unroll
  for (int r = 0; r < 8; ++r) {
    const int row = m0 + r + ((lane & 16) ? 8 : 0);
    const long idx = (long)row * N + ncol;
    if constexpr (OBF) ((bf16_t*)outp)[idx] = f2bf(acc[r]);
    else ((float*)outp)[idx] = acc[r];
  }
}

// ---------------------------------------------------------------------------
// Attention step: e[t] = w_e . tanh(Ws_pre[b,t,:] + q[b,:]);  softmax over Tp;
// a[b,:] = sum_t alpha[t] * h_s[b,t,:].   One workgroup per batch row.
// ---------------------------------------------------------------------------
__global__ __launch_bounds__(256) void attn_kernel(
    const bf16_t* __restrict__ ws_pre,  // [B,Tp,H] bf16
    const bf16_t* __restrict__ h_s,     // [B,Tp,H] bf16
    const float* __restrict__ q,        // [B,H] fp32
    const float* __restrict__ w_e,      // [H]
    bf16_t* __restrict__ a_out) {       // [B,H] bf16
  __shared__ float qs[HH];
  __shared__ float wes[HH];
  __shared__ float ev[TP];
  __shared__ float invsum;
  const int b = blockIdx.x;
  for (int i = threadIdx.x; i < HH; i += 256) {
    qs[i] = q[(long)b * HH + i];
    wes[i] = w_e[i];
  }
  __syncthreads();

  const int lane = threadIdx.x & 31;
  const int wave = threadIdx.x >> 5;
  const bf16_t* wsb = ws_pre + (long)b * TP * HH;
#pragma unroll 1
  for (int ti = 0; ti < 16; ++ti) {
    const int t = wave * 16 + ti;
    float s = 0.0f;
    for (int h = lane; h < HH; h += 32)
      s += wes[h] * fast_tanh(bf2f(wsb[(long)t * HH + h]) + qs[h]);
#pragma unroll
    for (int o = 16; o > 0; o >>= 1) s += __shfl_xor(s, o, 32);
    if (lane == 0) ev[t] = s;
  }
  __syncthreads();

  if (threadIdx.x == 0) {
    float mx = ev[0];
    for (int t = 1; t < TP; ++t) mx = fmaxf(mx, ev[t]);
    float sum = 0.0f;
    for (int t = 0; t < TP; ++t) {
      const float p = __expf(ev[t] - mx);
      ev[t] = p;
      sum += p;
    }
    invsum = 1.0f / sum;
  }
  __syncthreads();

  const float inv = invsum;
  const bf16_t* hsb = h_s + (long)b * TP * HH;
  for (int h = threadIdx.x; h < HH; h += 256) {
    float acc = 0.0f;
    for (int t = 0; t < TP; ++t) acc += ev[t] * bf2f(hsb[(long)t * HH + h]);
    a_out[(long)b * HH + h] = f2bf(acc * inv);
  }
}

// ---------------------------------------------------------------------------
// Small utility kernels
// ---------------------------------------------------------------------------
// Concatenate two fp32 weight matrices along K into one bf16 [N][KP1+K2] matrix
// (zero-padding src1 from K1 to KP1).
__global__ void wcat_kernel(bf16_t* __restrict__ dst,
                            const float* __restrict__ s1, int K1, int KP1,
                            const float* __restrict__ s2, int K2, int N) {
  const int KT = KP1 + K2;
  const long idx = (long)blockIdx.x * 256 + threadIdx.x;
  if (idx >= (long)N * KT) return;
  const int n = (int)(idx / KT);
  const int k = (int)(idx % KT);
  float v;
  if (k < KP1) v = (k < K1) ? s1[(long)n * K1 + k] : 0.0f;
  else         v = s2[(long)n * K2 + (k - KP1)];
  dst[idx] = f2bf(v);
}

__global__ void bias_kernel(float* __restrict__ dst, const float* __restrict__ b1,
                            const float* __restrict__ b2, int n) {
  const int i = blockIdx.x * 256 + threadIdx.x;
  if (i < n) dst[i] = b1[i] + b2[i];
}

// Embedding gather + bf16 convert + pad E=300 -> 320.
__global__ void embed_kernel(bf16_t* __restrict__ dst, const int* __restrict__ tok,
                             const float* __restrict__ tab, long NT) {
  const long idx = (long)blockIdx.x * 256 + threadIdx.x;
  if (idx >= NT * EP) return;
  const long r = idx / EP;
  const int k = (int)(idx % EP);
  const float v = (k < EE) ? tab[(long)tok[r] * EE + k] : 0.0f;
  dst[idx] = f2bf(v);
}

__global__ void zero_kernel(unsigned int* __restrict__ p, long n) {
  const long i = (long)blockIdx.x * 256 + threadIdx.x;
  if (i < n) p[i] = 0u;
}

__global__ void fc_kernel(const bf16_t* __restrict__ hm, const float* __restrict__ Wfc,
                          const float* __restrict__ bfc, float* __restrict__ out) {
  const int b = threadIdx.x;  // 128 threads
  float a0 = bfc[0], a1 = bfc[1], a2 = bfc[2];
  for (int h = 0; h < HH; ++h) {
    const float hv = bf2f(hm[(long)b * HH + h]);
    a0 += hv * Wfc[0 * HH + h];
    a1 += hv * Wfc[1 * HH + h];
    a2 += hv * Wfc[2 * HH + h];
  }
  out[b * 3 + 0] = a0;
  out[b * 3 + 1] = a1;
  out[b * 3 + 2] = a2;
}

// ---------------------------------------------------------------------------
// Host orchestration
// ---------------------------------------------------------------------------
extern "C" void kernel_launch(void* const* d_in, const int* in_sizes, int n_in,
                              void* d_out, int out_size, void* d_ws, size_t ws_size,
                              hipStream_t stream) {
  const int*   premise = (const int*)d_in[0];
  const int*   plen    = (const int*)d_in[1];
  const int*   hypoth  = (const int*)d_in[2];
  const int*   hlen    = (const int*)d_in[3];
  const float* table   = (const float*)d_in[4];
  const float* Wih_p = (const float*)d_in[5],  *Whh_p = (const float*)d_in[6];
  const float* bih_p = (const float*)d_in[7],  *bhh_p = (const float*)d_in[8];
  const float* Wih_h = (const float*)d_in[9],  *Whh_h = (const float*)d_in[10];
  const float* bih_h = (const float*)d_in[11], *bhh_h = (const float*)d_in[12];
  const float* Wih_m = (const float*)d_in[13], *Whh_m = (const float*)d_in[14];
  const float* bih_m = (const float*)d_in[15], *bhh_m = (const float*)d_in[16];
  const float* w_e  = (const float*)d_in[17];
  const float* W_s  = (const float*)d_in[18];
  const float* W_t  = (const float*)d_in[19];
  const float* W_m  = (const float*)d_in[20];
  const float* W_fc = (const float*)d_in[21];
  const float* b_fc = (const float*)d_in[22];
  float* out = (float*)d_out;

  // ---- workspace carve ----
  char* base = (char*)d_ws;
  size_t off = 0;
  auto carve = [&](size_t bytes) -> void* {
    void* p = base + off;
    off = (off + bytes + 255) & ~(size_t)255;
    return p;
  };
  bf16_t* emb_p  = (bf16_t*)carve((size_t)BB * TP * EP * 2);
  bf16_t* emb_h  = (bf16_t*)carve((size_t)BB * THL * EP * 2);
  bf16_t* Wp_cat = (bf16_t*)carve((size_t)G4 * (EP + HH) * 2);
  bf16_t* Wh_cat = (bf16_t*)carve((size_t)G4 * (EP + HH) * 2);
  bf16_t* Wm_cat = (bf16_t*)carve((size_t)G4 * (3 * HH) * 2);
  bf16_t* Wq_cat = (bf16_t*)carve((size_t)HH * (2 * HH) * 2);
  bf16_t* Ws_bf  = (bf16_t*)carve((size_t)HH * HH * 2);
  float*  bias_p = (float*)carve((size_t)G4 * 4);
  float*  bias_h = (float*)carve((size_t)G4 * 4);
  float*  bias_m = (float*)carve((size_t)G4 * 4);
  bf16_t* h_s    = (bf16_t*)carve((size_t)BB * TP * HH * 2);
  bf16_t* ws_pre = (bf16_t*)carve((size_t)BB * TP * HH * 2);
  bf16_t* h_t    = (bf16_t*)carve((size_t)BB * THL * HH * 2);
  bf16_t* hA     = (bf16_t*)carve((size_t)BB * HH * 2);
  bf16_t* hB     = (bf16_t*)carve((size_t)BB * HH * 2);
  float*  c_st   = (float*)carve((size_t)BB * HH * 4);
  bf16_t* hmA    = (bf16_t*)carve((size_t)BB * HH * 2);
  bf16_t* hmB    = (bf16_t*)carve((size_t)BB * HH * 2);
  float*  cm_st  = (float*)carve((size_t)BB * HH * 4);
  bf16_t* a_buf  = (bf16_t*)carve((size_t)BB * HH * 2);
  float*  q_f32  = (float*)carve((size_t)BB * HH * 4);

  auto zero32 = [&](void* p, long n32) {
    zero_kernel<<<dim3((unsigned)((n32 + 255) / 256)), 256, 0, stream>>>(
        (unsigned int*)p, n32);
  };
  auto wcat = [&](bf16_t* dst, const float* s1, int K1, int KP1, const float* s2,
                  int K2, int N) {
    const long tot = (long)N * (KP1 + K2);
    wcat_kernel<<<dim3((unsigned)((tot + 255) / 256)), 256, 0, stream>>>(
        dst, s1, K1, KP1, s2, K2, N);
  };

  // ---- phase 0: weight prep (fp32 -> bf16, K-concat, bias add) ----
  wcat(Wp_cat, Wih_p, EE, EP, Whh_p, HH, G4);        // [2048][832]
  wcat(Wh_cat, Wih_h, EE, EP, Whh_h, HH, G4);        // [2048][832]
  wcat(Wm_cat, Wih_m, 2 * HH, 2 * HH, Whh_m, HH, G4);// [2048][1536]
  wcat(Wq_cat, W_t, HH, HH, W_m, HH, HH);            // [512][1024]
  wcat(Ws_bf, W_s, HH, HH, nullptr, 0, HH);          // [512][512]
  bias_kernel<<<dim3(G4 / 256), 256, 0, stream>>>(bias_p, bih_p, bhh_p, G4);
  bias_kernel<<<dim3(G4 / 256), 256, 0, stream>>>(bias_h, bih_h, bhh_h, G4);
  bias_kernel<<<dim3(G4 / 256), 256, 0, stream>>>(bias_m, bih_m, bhh_m, G4);

  // ---- phase 1: embedding gather (bf16, padded) ----
  {
    const long nt_p = (long)BB * TP, nt_h = (long)BB * THL;
    embed_kernel<<<dim3((unsigned)((nt_p * EP + 255) / 256)), 256, 0, stream>>>(
        emb_p, premise, table, nt_p);
    embed_kernel<<<dim3((unsigned)((nt_h * EP + 255) / 256)), 256, 0, stream>>>(
        emb_h, hypoth, table, nt_h);
  }

  // ---- phase 2: premise LSTM (Tp fused WMMA steps) ----
  zero32(hA, (long)BB * HH / 2);
  zero32(c_st, (long)BB * HH);
  for (int t = 0; t < TP; ++t) {
    const bf16_t* hin = (t & 1) ? hB : hA;
    bf16_t* hout = (t & 1) ? hA : hB;
    rnn_step_kernel<EP, HH, 0><<<dim3(32), 256, 0, stream>>>(
        emb_p + (long)t * EP, (long)TP * EP, hin, (long)HH, nullptr, 0,
        Wp_cat, bias_p, c_st, hin, hout, h_s + (long)t * HH, (long)TP * HH,
        plen, t);
  }

  // ---- phase 3: hypothesis LSTM (Th fused WMMA steps) ----
  zero32(hA, (long)BB * HH / 2);
  zero32(c_st, (long)BB * HH);
  for (int t = 0; t < THL; ++t) {
    const bf16_t* hin = (t & 1) ? hB : hA;
    bf16_t* hout = (t & 1) ? hA : hB;
    rnn_step_kernel<EP, HH, 0><<<dim3(32), 256, 0, stream>>>(
        emb_h + (long)t * EP, (long)THL * EP, hin, (long)HH, nullptr, 0,
        Wh_cat, bias_h, c_st, hin, hout, h_t + (long)t * HH, (long)THL * HH,
        hlen, t);
  }

  // ---- phase 4: Ws_pre = h_s @ W_s^T  (one big batched WMMA GEMM) ----
  gemm_kernel<HH, 0, true><<<dim3(4096), 256, 0, stream>>>(
      h_s, (long)HH, nullptr, 0, Ws_bf, ws_pre, BB * TP, HH);

  // ---- phase 5: match-LSTM loop with attention ----
  zero32(hmA, (long)BB * HH / 2);
  zero32(cm_st, (long)BB * HH);
  for (int k = 0; k < THL; ++k) {
    const bf16_t* hm_in = (k & 1) ? hmB : hmA;
    bf16_t* hm_out = (k & 1) ? hmA : hmB;
    // q = htk @ W_t^T + h_m @ W_m^T
    gemm_kernel<HH, HH, false><<<dim3(32), 256, 0, stream>>>(
        h_t + (long)k * HH, (long)THL * HH, hm_in, (long)HH, Wq_cat, q_f32,
        BB, HH);
    // attention over premise -> a
    attn_kernel<<<dim3(BB), 256, 0, stream>>>(ws_pre, h_s, q_f32, w_e, a_buf);
    // LSTMCell([a | htk | h_m])
    rnn_step_kernel<HH, HH, HH><<<dim3(32), 256, 0, stream>>>(
        a_buf, (long)HH, h_t + (long)k * HH, (long)THL * HH, hm_in, (long)HH,
        Wm_cat, bias_m, cm_st, hm_in, hm_out, nullptr, 0, hlen, k);
  }
  // after 64 steps the final h_m lives in hmA (even step count)

  // ---- phase 6: FC head ----
  fc_kernel<<<dim3(1), 128, 0, stream>>>(hmA, W_fc, b_fc, out);
}
